// BackmapGNN_81561428951046
// MI455X (gfx1250) — compile-verified, matrix operand order checked by reading, not run on hardware
//
#include <hip/hip_runtime.h>

// ---------------------------------------------------------------------------
// BackmapGNN on MI455X (gfx1250): wave32 WMMA f16 (f32 accum) pipeline.
// Each wave: 32 rows x 64 cols (8 accumulators), B fragments reused 2x.
// ---------------------------------------------------------------------------

#define N_NODES   8192
#define N_EDGES   131072
#define N_ATOMS   6144
#define HID       256
#define NLAYERS   6
#define K_IN      416     // 400 padded to multiple of 32
#define K_GEOM    64      // 51 padded
#define K_NODE    512

typedef _Float16 f16;
typedef _Float16 f16x8  __attribute__((ext_vector_type(8)));
typedef _Float16 f16x16 __attribute__((ext_vector_type(16)));
typedef float    f32x8  __attribute__((ext_vector_type(8)));

__device__ __forceinline__ float silu_f(float x) { return x / (1.0f + __expf(-x)); }

__device__ __forceinline__ void atom_add_f32(float* p, float v) {
    __hip_atomic_fetch_add(p, v, __ATOMIC_RELAXED, __HIP_MEMORY_SCOPE_AGENT);
}

// Load a 16x32 f16 WMMA fragment slice for this lane:
// lane L: m/n = L&15, halfsel = L>>4; elements 0..7 at k = halfsel*8 + i,
// elements 8..15 at k = 16 + halfsel*8 + (i-8).  Two 16B vector loads.
__device__ __forceinline__ f16x16 load_frag(const f16* __restrict__ p) {
    f16x8 lo = *(const f16x8*)p;         // k = halfsel*8 .. +7
    f16x8 hi = *(const f16x8*)(p + 16);  // k = 16 + halfsel*8 .. +7
    return __builtin_shufflevector(lo, hi, 0,1,2,3,4,5,6,7,8,9,10,11,12,13,14,15);
}

// ---------------------------------------------------------------------------
// Generic WMMA GEMM:  C[M,256] = A[M,K] @ Bt^T  (+bias, epilogue by MODE)
//   Bt is pre-transposed weights: Bt[n*K + k], K % 32 == 0.
//   Block = 256 threads = 8 waves; block covers 64 rows x 256 cols.
//   Wave w: rows (w>>2)*32 (two 16-row strips), col group (w&3)*64.
//   8 accumulators: acc[2*j + strip], B fragment j reused by both strips.
// MODE 0: C16[m*ldc+n] = v
// MODE 1: C16[m*ldc+n] = silu(v)
// MODE 2: atomicAdd(agg[dstIdx[m]*256+n], v*keep[m])   (fused segment_sum)
// MODE 3: h = Hf[m*256+n]+v; Hf=h; H16[m*ldh+n]=f16(h) (fused residual)
// ---------------------------------------------------------------------------
template<int MODE>
__global__ void __launch_bounds__(256)
k_gemm(const f16* __restrict__ A, int lda,
       const int* __restrict__ rowIdx,
       const f16* __restrict__ Bt, int K,
       const float* __restrict__ bias,
       f16* __restrict__ C16, int ldc,
       const float* __restrict__ keep,
       const int* __restrict__ dstIdx,
       float* __restrict__ Hf,
       f16* __restrict__ H16, int ldh)
{
    const int wave     = threadIdx.x >> 5;
    const int lane     = threadIdx.x & 31;
    const int hsel     = lane >> 4;
    const int nl       = lane & 15;
    const int rowBase0 = blockIdx.x * 64 + (wave >> 2) * 32;
    const int rowBase1 = rowBase0 + 16;
    const int colBase  = (wave & 3) * 64;

    // A fragment rows for this lane (both lane halves read the same row).
    const int am0 = rowBase0 + nl;
    const int am1 = rowBase1 + nl;
    const long arow0 = rowIdx ? (long)rowIdx[am0] : (long)am0;
    const long arow1 = rowIdx ? (long)rowIdx[am1] : (long)am1;
    const f16* Ap0 = A  + arow0 * (long)lda + hsel * 8;
    const f16* Ap1 = A  + arow1 * (long)lda + hsel * 8;
    const f16* Bp  = Bt + (long)(colBase + nl) * K + hsel * 8;
    const long bstride = (long)16 * K;

    f32x8 acc[8] = {};
    for (int k0 = 0; k0 < K; k0 += 32) {
        f16x16 a0 = load_frag(Ap0 + k0);
        f16x16 a1 = load_frag(Ap1 + k0);
        #pragma unroll
        for (int j = 0; j < 4; ++j) {
            f16x16 b = load_frag(Bp + j * bstride + k0);
            acc[2 * j]     = __builtin_amdgcn_wmma_f32_16x16x32_f16(false, a0, false, b, (short)0, acc[2 * j],     false, false);
            acc[2 * j + 1] = __builtin_amdgcn_wmma_f32_16x16x32_f16(false, a1, false, b, (short)0, acc[2 * j + 1], false, false);
        }
    }

    #pragma unroll
    for (int s = 0; s < 2; ++s) {
        #pragma unroll
        for (int r = 0; r < 8; ++r) {
            const int m = (s ? rowBase1 : rowBase0) + r + hsel * 8;  // C layout: VGPR r -> M=r / M=r+8
            const float kv  = (MODE == 2) ? keep[m] : 0.0f;
            const long drow = (MODE == 2) ? (long)dstIdx[m] * HID : 0;
            #pragma unroll
            for (int j = 0; j < 4; ++j) {
                const int n = colBase + j * 16 + nl;
                float v = acc[2 * j + s][r] + (bias ? bias[n] : 0.0f);
                if (MODE == 0) {
                    C16[(long)m * ldc + n] = (f16)v;
                } else if (MODE == 1) {
                    C16[(long)m * ldc + n] = (f16)silu_f(v);
                } else if (MODE == 2) {
                    atom_add_f32(&Hf[drow + n], v * kv);
                } else {
                    const long o = (long)m * HID + n;
                    const float hn = Hf[o] + v;
                    Hf[o] = hn;
                    H16[(long)m * ldh + n] = (f16)hn;
                }
            }
        }
    }
}

// --------------------------- weight convert/transpose ----------------------
// dst[n*Kp + k] = f16( src[(k+rowOff)*Ncols + n] ), zero-padded for k>=Ksrc
__global__ void k_wconv(const float* __restrict__ src, f16* __restrict__ dst,
                        int Ksrc, int Kp, int Ncols, int rowOff)
{
    int idx = blockIdx.x * blockDim.x + threadIdx.x;
    if (idx >= Ncols * Kp) return;
    int n = idx / Kp, k = idx - n * Kp;
    float v = (k < Ksrc) ? src[(long)(k + rowOff) * Ncols + n] : 0.0f;
    dst[idx] = (f16)v;
}

__global__ void k_zero(float* __restrict__ p, int count)
{
    int i = blockIdx.x * blockDim.x + threadIdx.x;
    if (i < count) p[i] = 0.0f;
}

// --------------------------- node features [N, 416] f16 --------------------
__global__ void k_nodefeat(const int* __restrict__ ntype, const int* __restrict__ nname,
                           const int* __restrict__ nresname, const int* __restrict__ nrif,
                           const int* __restrict__ nag, const float* __restrict__ t_node,
                           const float* __restrict__ geom,
                           const float* __restrict__ emb_nt, const float* __restrict__ emb_name,
                           const float* __restrict__ emb_rn, const float* __restrict__ emb_ag,
                           const float* __restrict__ gp_w, const float* __restrict__ gp_b,
                           f16* __restrict__ x)
{
    int i = blockIdx.x * blockDim.x + threadIdx.x;
    if (i >= N_NODES) return;
    f16* xp = x + (long)i * K_IN;
    int o = 0;
    const float* e = emb_nt + ntype[i] * 32;
    for (int c = 0; c < 32; ++c) xp[o++] = (f16)e[c];
    e = emb_name + nname[i] * 64;
    for (int c = 0; c < 64; ++c) xp[o++] = (f16)e[c];
    e = emb_rn + nresname[i] * 64;
    for (int c = 0; c < 64; ++c) xp[o++] = (f16)e[c];
    e = emb_ag + nag[i] * 16;
    for (int c = 0; c < 16; ++c) xp[o++] = (f16)e[c];
    // sinusoidal(res_in_frame, 64)
    {
        float v = (float)nrif[i];
        for (int j = 0; j < 32; ++j) {
            float fr = __expf(-9.2103403719761836f * (float)j / 32.0f);
            float a = v * fr;
            xp[o + j]      = (f16)__sinf(a);
            xp[o + 32 + j] = (f16)__cosf(a);
        }
        o += 64;
    }
    // sinusoidal(t, 128)
    {
        float v = t_node[i];
        for (int j = 0; j < 64; ++j) {
            float fr = __expf(-9.2103403719761836f * (float)j / 64.0f);
            float a = v * fr;
            xp[o + j]      = (f16)__sinf(a);
            xp[o + 64 + j] = (f16)__cosf(a);
        }
        o += 128;
    }
    // geom_proj: [5] @ [5,32] + b
    for (int c = 0; c < 32; ++c) {
        float s = gp_b[c];
        for (int d = 0; d < 5; ++d) s += geom[i * 5 + d] * gp_w[d * 32 + c];
        xp[o++] = (f16)s;
    }
    for (; o < K_IN; ++o) xp[o] = (f16)0.0f;   // pad 400->416
}

// --------------------------- edge geometry -> g[E,64] f16, keep[E] ---------
__global__ void k_edgegeom(const int* __restrict__ ei, const int* __restrict__ etype,
                           const float* __restrict__ pos, const int* __restrict__ nres,
                           const float* __restrict__ bb, const float* __restrict__ emb_et,
                           f16* __restrict__ g, float* __restrict__ keep)
{
    int eidx = blockIdx.x * blockDim.x + threadIdx.x;
    if (eidx >= N_EDGES) return;
    int s = ei[eidx], d = ei[N_EDGES + eidx];
    float r0 = pos[s * 3 + 0] - pos[d * 3 + 0];
    float r1 = pos[s * 3 + 1] - pos[d * 3 + 1];
    float r2 = pos[s * 3 + 2] - pos[d * 3 + 2];
    float d2 = r0 * r0 + r1 * r1 + r2 * r2;
    float dist = sqrtf(fmaxf(d2, 1e-8f));
    int t = etype[eidx];
    bool is_bb  = (t == 0);
    bool is_int = (t == 3) || (t == 4);
    bool kp = (!is_bb || dist <= 10.0f) && (!is_int || dist <= 10.0f);
    keep[eidx] = kp ? 1.0f : 0.0f;
    const float* R = bb + (long)nres[d] * 9;
    float inv = 1.0f / dist;
    f16* gp = g + (long)eidx * K_GEOM;
    // rbf: centers linspace(0,20,16), width 20/16
    for (int j = 0; j < 16; ++j) {
        float c = 20.0f * (float)j / 15.0f;
        float z = (dist - c) * 0.8f;      // 1/1.25
        gp[j] = (f16)__expf(-z * z);
    }
    // dir_local = (R^T @ rel) / dist
    for (int j = 0; j < 3; ++j) {
        float rl = R[0 * 3 + j] * r0 + R[1 * 3 + j] * r1 + R[2 * 3 + j] * r2;
        gp[16 + j] = (f16)(rl * inv);
    }
    int tt = t < 0 ? 0 : (t > 7 ? 7 : t);
    const float* et = emb_et + tt * 32;
    for (int j = 0; j < 32; ++j) gp[19 + j] = (f16)et[j];
    for (int j = 51; j < K_GEOM; ++j) gp[j] = (f16)0.0f;
}

// ------------- e1 = silu(hs[src] + hd[dst] + cpart)  [E,256] f16 -----------
__global__ void k_edge_assemble(const int* __restrict__ ei,
                                const f16* __restrict__ hs, const f16* __restrict__ hd,
                                const f16* __restrict__ cpart, f16* __restrict__ e1)
{
    int idx = blockIdx.x * blockDim.x + threadIdx.x;   // E*32 threads, 8 halves each
    int e = idx >> 5;
    int c8 = (idx & 31) * 8;
    if (e >= N_EDGES) return;
    int s = ei[e], d = ei[N_EDGES + e];
    f16x8 a = *(const f16x8*)(hs + (long)s * HID + c8);
    f16x8 b = *(const f16x8*)(hd + (long)d * HID + c8);
    f16x8 c = *(const f16x8*)(cpart + (long)e * HID + c8);
    f16x8 o;
    #pragma unroll
    for (int k = 0; k < 8; ++k) {
        float v = (float)a[k] + (float)b[k] + (float)c[k];
        o[k] = (f16)silu_f(v);
    }
    *(f16x8*)(e1 + (long)e * HID + c8) = o;
}

// ------------- nin[:,256:512] = f16(agg)  ----------------------------------
__global__ void k_agg16(const float* __restrict__ agg, f16* __restrict__ nin)
{
    int idx = blockIdx.x * blockDim.x + threadIdx.x;   // N*32 threads
    int n = idx >> 5;
    int c8 = (idx & 31) * 8;
    if (n >= N_NODES) return;
    f16x8 o;
    #pragma unroll
    for (int k = 0; k < 8; ++k) o[k] = (f16)agg[(long)n * HID + c8 + k];
    *(f16x8*)(nin + (long)n * K_NODE + HID + c8) = o;
}

// ------------- output head: [NA,256]f16 @ [256,5] + spherical --------------
__global__ void k_outhead(const f16* __restrict__ r2, const float* __restrict__ Wo2,
                          const float* __restrict__ bo2, float* __restrict__ out)
{
    int a = blockIdx.x * blockDim.x + threadIdx.x;
    if (a >= N_ATOMS) return;
    float acc[5];
    for (int q = 0; q < 5; ++q) acc[q] = bo2[q];
    const f16* hp = r2 + (long)a * HID;
    for (int c = 0; c < HID; ++c) {
        float hv = (float)hp[c];
        #pragma unroll
        for (int q = 0; q < 5; ++q) acc[q] += hv * Wo2[c * 5 + q];
    }
    float r   = 6.0f / (1.0f + __expf(-acc[0]));
    float a2  = fabsf(acc[2]);
    float n1v = sqrtf(fmaxf(acc[1] * acc[1] + a2 * a2, 1e-8f));
    float ct = acc[1] / n1v, st = a2 / n1v;
    float n2v = sqrtf(fmaxf(acc[3] * acc[3] + acc[4] * acc[4], 1e-8f));
    float cp = acc[3] / n2v, sp = acc[4] / n2v;
    out[a * 5 + 0] = r;  out[a * 5 + 1] = ct; out[a * 5 + 2] = st;
    out[a * 5 + 3] = cp; out[a * 5 + 4] = sp;
    float* pl = out + N_ATOMS * 5 + a * 3;
    pl[0] = r * st * cp; pl[1] = r * st * sp; pl[2] = r * ct;
}

// ===========================================================================
extern "C" void kernel_launch(void* const* d_in, const int* in_sizes, int n_in,
                              void* d_out, int out_size, void* d_ws, size_t ws_size,
                              hipStream_t stream) {
    (void)in_sizes; (void)n_in; (void)out_size; (void)ws_size;
    const float* node_pos   = (const float*)d_in[0];
    const int*   node_type  = (const int*)d_in[1];
    const int*   node_name  = (const int*)d_in[2];
    const int*   node_rn    = (const int*)d_in[3];
    const int*   node_res   = (const int*)d_in[4];
    const int*   node_rif   = (const int*)d_in[5];
    const int*   node_ag    = (const int*)d_in[6];
    const int*   edge_index = (const int*)d_in[7];
    const int*   edge_type  = (const int*)d_in[8];
    const float* bb_frames  = (const float*)d_in[9];
    const float* t_node     = (const float*)d_in[10];
    const float* geom_sph   = (const float*)d_in[11];
    const int*   atom_idx   = (const int*)d_in[12];
    const float* emb_nt     = (const float*)d_in[13];
    const float* emb_name   = (const float*)d_in[14];
    const float* emb_rn     = (const float*)d_in[15];
    const float* emb_ag     = (const float*)d_in[16];
    const float* emb_et     = (const float*)d_in[17];
    const float* gp_w = (const float*)d_in[18];
    const float* gp_b = (const float*)d_in[19];
    const float* ip_w = (const float*)d_in[20];
    const float* ip_b = (const float*)d_in[21];
    const float* ew0 = (const float*)d_in[22]; const float* eb0 = (const float*)d_in[23];
    const float* ew1 = (const float*)d_in[24]; const float* eb1 = (const float*)d_in[25];
    const float* ew2 = (const float*)d_in[26]; const float* eb2 = (const float*)d_in[27];
    const float* nw0 = (const float*)d_in[28]; const float* nb0 = (const float*)d_in[29];
    const float* nw1 = (const float*)d_in[30]; const float* nb1 = (const float*)d_in[31];
    const float* nw2 = (const float*)d_in[32]; const float* nb2 = (const float*)d_in[33];
    const float* ow0 = (const float*)d_in[34]; const float* ob0 = (const float*)d_in[35];
    const float* ow1 = (const float*)d_in[36]; const float* ob1 = (const float*)d_in[37];
    const float* ow2 = (const float*)d_in[38]; const float* ob2 = (const float*)d_in[39];

    char* ws = (char*)d_ws;
    size_t off = 0;
    auto alloc = [&](size_t bytes) -> void* {
        off = (off + 255) & ~(size_t)255;
        void* p = ws + off; off += bytes; return p;
    };
    f16* Wi_t   = (f16*)alloc((size_t)HID * K_IN * 2);
    f16* We0a_t = (f16*)alloc((size_t)HID * HID * 2);
    f16* We0b_t = (f16*)alloc((size_t)HID * HID * 2);
    f16* Wc_t   = (f16*)alloc((size_t)HID * K_GEOM * 2);
    f16* We1_t  = (f16*)alloc((size_t)HID * HID * 2);
    f16* We2_t  = (f16*)alloc((size_t)HID * HID * 2);
    f16* Wn0_t  = (f16*)alloc((size_t)HID * K_NODE * 2);
    f16* Wn1_t  = (f16*)alloc((size_t)HID * HID * 2);
    f16* Wn2_t  = (f16*)alloc((size_t)HID * HID * 2);
    f16* Wo0_t  = (f16*)alloc((size_t)HID * HID * 2);
    f16* Wo1_t  = (f16*)alloc((size_t)HID * HID * 2);
    f16*   x16  = (f16*)alloc((size_t)N_NODES * K_IN * 2);
    float* hbuf = (float*)alloc((size_t)N_NODES * HID * 4);
    f16*   nin  = (f16*)alloc((size_t)N_NODES * K_NODE * 2); // [h16 | agg16]
    f16*   hs   = (f16*)alloc((size_t)N_NODES * HID * 2);
    f16*   hd   = (f16*)alloc((size_t)N_NODES * HID * 2);
    f16*   g    = (f16*)alloc((size_t)N_EDGES * K_GEOM * 2);
    f16*   cpart= (f16*)alloc((size_t)N_EDGES * HID * 2);
    float* keep = (float*)alloc((size_t)N_EDGES * 4);
    f16*   e1   = (f16*)alloc((size_t)N_EDGES * HID * 2);
    f16*   e2   = (f16*)alloc((size_t)N_EDGES * HID * 2);
    float* agg  = (float*)alloc((size_t)N_NODES * HID * 4);
    f16*   n1   = (f16*)alloc((size_t)N_NODES * HID * 2);
    f16*   n2   = (f16*)alloc((size_t)N_NODES * HID * 2);
    f16*   r1   = (f16*)alloc((size_t)N_ATOMS * HID * 2);
    f16*   r2b  = (f16*)alloc((size_t)N_ATOMS * HID * 2);

    auto wconv = [&](const float* src, f16* dst, int Ksrc, int Kp, int rowOff) {
        int total = HID * Kp;
        k_wconv<<<(total + 255) / 256, 256, 0, stream>>>(src, dst, Ksrc, Kp, HID, rowOff);
    };
    wconv(ip_w, Wi_t,  400, K_IN,   0);
    wconv(ew0,  We0a_t,256, HID,    0);
    wconv(ew0,  We0b_t,256, HID,    256);
    wconv(ew0,  Wc_t,  51,  K_GEOM, 512);
    wconv(ew1,  We1_t, 256, HID,    0);
    wconv(ew2,  We2_t, 256, HID,    0);
    wconv(nw0,  Wn0_t, 512, K_NODE, 0);
    wconv(nw1,  Wn1_t, 256, HID,    0);
    wconv(nw2,  Wn2_t, 256, HID,    0);
    wconv(ow0,  Wo0_t, 256, HID,    0);
    wconv(ow1,  Wo1_t, 256, HID,    0);

    // node features + input projection (residual-init into zeroed h)
    k_nodefeat<<<N_NODES / 256, 256, 0, stream>>>(node_type, node_name, node_rn, node_rif,
                                                  node_ag, t_node, geom_sph,
                                                  emb_nt, emb_name, emb_rn, emb_ag,
                                                  gp_w, gp_b, x16);
    k_zero<<<(N_NODES * HID) / 256, 256, 0, stream>>>(hbuf, N_NODES * HID);
    k_gemm<3><<<N_NODES / 64, 256, 0, stream>>>(x16, K_IN, nullptr, Wi_t, K_IN, ip_b,
                                                nullptr, 0, nullptr, nullptr, hbuf, nin, K_NODE);

    // edge geometry + layer-invariant cpart = g @ W0c + b_e0  (f16)
    k_edgegeom<<<N_EDGES / 256, 256, 0, stream>>>(edge_index, edge_type, node_pos, node_res,
                                                  bb_frames, emb_et, g, keep);
    k_gemm<0><<<N_EDGES / 64, 256, 0, stream>>>(g, K_GEOM, nullptr, Wc_t, K_GEOM, eb0,
                                                cpart, HID, nullptr, nullptr, nullptr, nullptr, 0);

    const int* dst_idx = edge_index + N_EDGES;
    for (int layer = 0; layer < NLAYERS; ++layer) {
        // hs = h16 @ W0a,  hd = h16 @ W0b   (node-level split of edge GEMM-1)
        k_gemm<0><<<N_NODES / 64, 256, 0, stream>>>(nin, K_NODE, nullptr, We0a_t, HID, nullptr,
                                                    hs, HID, nullptr, nullptr, nullptr, nullptr, 0);
        k_gemm<0><<<N_NODES / 64, 256, 0, stream>>>(nin, K_NODE, nullptr, We0b_t, HID, nullptr,
                                                    hd, HID, nullptr, nullptr, nullptr, nullptr, 0);
        // e1 = silu(hs[src] + hd[dst] + cpart)
        k_edge_assemble<<<(N_EDGES * 32) / 256, 256, 0, stream>>>(edge_index, hs, hd, cpart, e1);
        // e2 = silu(e1 @ We1 + b1)
        k_gemm<1><<<N_EDGES / 64, 256, 0, stream>>>(e1, HID, nullptr, We1_t, HID, eb1,
                                                    e2, HID, nullptr, nullptr, nullptr, nullptr, 0);
        // agg = segment_sum((e2 @ We2 + b2) * keep, dst)  (fused atomic epilogue)
        k_zero<<<(N_NODES * HID) / 256, 256, 0, stream>>>(agg, N_NODES * HID);
        k_gemm<2><<<N_EDGES / 64, 256, 0, stream>>>(e2, HID, nullptr, We2_t, HID, eb2,
                                                    nullptr, 0, keep, dst_idx, agg, nullptr, 0);
        // node MLP on [h | agg] with fused residual update of h / h16
        k_agg16<<<(N_NODES * 32) / 256, 256, 0, stream>>>(agg, nin);
        k_gemm<1><<<N_NODES / 64, 256, 0, stream>>>(nin, K_NODE, nullptr, Wn0_t, K_NODE, nb0,
                                                    n1, HID, nullptr, nullptr, nullptr, nullptr, 0);
        k_gemm<1><<<N_NODES / 64, 256, 0, stream>>>(n1, HID, nullptr, Wn1_t, HID, nb1,
                                                    n2, HID, nullptr, nullptr, nullptr, nullptr, 0);
        k_gemm<3><<<N_NODES / 64, 256, 0, stream>>>(n2, HID, nullptr, Wn2_t, HID, nb2,
                                                    nullptr, 0, nullptr, nullptr, hbuf, nin, K_NODE);
    }

    // output head: gather atoms via rowIdx directly in the GEMM A-load
    k_gemm<1><<<N_ATOMS / 64, 256, 0, stream>>>(nin, K_NODE, atom_idx, Wo0_t, HID, ob0,
                                                r1, HID, nullptr, nullptr, nullptr, nullptr, 0);
    k_gemm<1><<<N_ATOMS / 64, 256, 0, stream>>>(r1, HID, nullptr, Wo1_t, HID, ob1,
                                                r2b, HID, nullptr, nullptr, nullptr, nullptr, 0);
    k_outhead<<<N_ATOMS / 256, 256, 0, stream>>>(r2b, ow2, ob2, (float*)d_out);
}
